// LIDIA_84232898609599
// MI455X (gfx1250) — compile-verified, matrix operand order checked by reading, not run on hardware
//
#include <hip/hip_runtime.h>
#include <hip/hip_bf16.h>
#include <math.h>

typedef __attribute__((ext_vector_type(16))) _Float16 v16h;
typedef __attribute__((ext_vector_type(8)))  _Float16 v8h;
typedef __attribute__((ext_vector_type(8)))  float    v8f;

#define PS    5
#define KNN   14
#define WR    14
#define NOFF  29              // 2*WR+1
#define TT    2
#define CC    3
#define HH    192
#define WW    192
#define PADW  4               // ps-1
#define HP    200
#define WP    200
#define NH    196
#define NW    196
#define NQTOT (TT*NH*NW)
#define XT    16
#define TILESX 13             // ceil(196/16)
#define GSTRIDE 52            // LDS gram row stride (bank-conflict-free)

__device__ __forceinline__ int reflect_idx(int j, int n) {
  if (j < 0) j = -j;
  if (j >= n) j = 2*(n-1) - j;
  return j;
}
__device__ __forceinline__ int clampi(int v, int lo, int hi) {
  return v < lo ? lo : (v > hi ? hi : v);
}

// ---------------- K1: per (t,c) mean of normalized image ----------------
__global__ void k_means(const float* __restrict__ in, float* __restrict__ means) {
  int b = blockIdx.x;                       // t*C + c, 6 blocks
  const float* p = in + (size_t)b * HH * WW;
  float s = 0.f;
  for (int i = threadIdx.x; i < HH*WW; i += blockDim.x)
    s += (p[i] * (1.0f/255.0f) - 0.5f) * 2.0f;
  __shared__ float red[256];
  red[threadIdx.x] = s; __syncthreads();
  for (int off = 128; off > 0; off >>= 1) {
    if ((int)threadIdx.x < off) red[threadIdx.x] += red[threadIdx.x + off];
    __syncthreads();
  }
  if (threadIdx.x == 0) means[b] = red[0] / (float)(HH*WW);
}

// ------------- K2a: reflect pad + normalize + gray conversion -----------
__global__ void k_pad_gray(const float* __restrict__ in, const float* __restrict__ means,
                           float* __restrict__ xp, float* __restrict__ gray) {
  int idx = blockIdx.x * blockDim.x + threadIdx.x;
  if (idx >= TT*HP*WP) return;
  int t = idx / (HP*WP); int r = idx % (HP*WP); int hp = r / WP; int wp = r % WP;
  int hs = reflect_idx(hp - PADW, HH);
  int ws = reflect_idx(wp - PADW, WW);
  float v[3];
  #pragma unroll
  for (int c = 0; c < CC; ++c) {
    float raw = in[((size_t)(t*CC + c)*HH + hs)*WW + ws];
    float x = (raw*(1.0f/255.0f) - 0.5f)*2.0f - means[t*CC + c];
    xp[((size_t)(t*CC + c)*HP + hp)*WP + wp] = x;
    v[c] = x;
  }
  gray[((size_t)t*HP + hp)*WP + wp] = 0.2989f*v[0] + 0.587f*v[1] + 0.114f*v[2];
}

// ------- K2b: f16 gray patch vectors (25 dims, padded to 32) + norms ----
__global__ void k_patches(const float* __restrict__ gray, _Float16* __restrict__ Pg,
                          float* __restrict__ norms) {
  int idx = blockIdx.x * blockDim.x + threadIdx.x;
  if (idx >= NQTOT) return;
  int t = idx / (NH*NW); int r = idx % (NH*NW); int y = r / NW; int x = r % NW;
  _Float16* dst = Pg + (size_t)idx * 32;
  float n = 0.f;
  int pp = 0;
  #pragma unroll
  for (int py = 0; py < PS; ++py)
    #pragma unroll
    for (int px = 0; px < PS; ++px, ++pp) {
      float g = gray[((size_t)t*HP + (y+py))*WP + (x+px)];
      _Float16 h = (_Float16)g;
      dst[pp] = h;
      float f = (float)h;
      n += f*f;
    }
  #pragma unroll
  for (int q = 25; q < 32; ++q) dst[q] = (_Float16)0.f;
  norms[idx] = n;
}

// -------- K3: WMMA banded-Gram patch search + streaming top-14 ----------
// One wave per (t, y, 16-query x-tile). Per row offset dy: prefetch next
// candidate row, branch-free norm staging, preload all three B fragments
// (6x b128 in flight), 3x v_wmma_f32_16x16x32_f16 produce the 16x48 Gram
// tile, D fragments scattered branch-free to LDS (stride 52 => conflict-
// free). Then 16 lanes compute dist = nq + nc - 2*dot over the valid
// 29-wide band and run a register-resident replace-worst top-14.
__global__ __launch_bounds__(32) void k_knn(const _Float16* __restrict__ Pg,
                                            const float* __restrict__ norms,
                                            float* __restrict__ tkD, int* __restrict__ tkI) {
  const int lane = threadIdx.x;
  const int bx = blockIdx.x;
  int t  = bx / (NH*TILESX);
  int r  = bx % (NH*TILESX);
  int y  = r / TILESX;
  int x0 = (r % TILESX) * XT;

  const int n16 = lane & 15;
  const int hi  = lane >> 4;        // 0 or 1
  const int k0  = hi * 8;           // f16 A/B fragment: runs [k0,k0+8) and [k0+16,k0+24)

  // A fragment: query M = n16 at row y (clamp x for partial tiles; ignored later)
  int qx = x0 + n16; if (qx > NW-1) qx = NW-1;
  const _Float16* arow = Pg + ((size_t)((t*NH + y)*NW) + qx) * 32;
  v8h alo = *(const v8h*)(arow + k0);
  v8h ahi = *(const v8h*)(arow + k0 + 16);
  v16h afrag;
  #pragma unroll
  for (int e = 0; e < 8; ++e) { afrag[e] = alo[e]; afrag[e+8] = ahi[e]; }

  __shared__ float s_nq[16];
  __shared__ float s_nc[64];
  __shared__ float s_gram[16 * GSTRIDE];
  if (lane < 16) s_nq[lane] = norms[(size_t)((t*NH + y)*NW) + qx];
  __syncthreads();

  float kd[KNN]; int ki[KNN];
  #pragma unroll
  for (int s = 0; s < KNN; ++s) { kd[s] = 3.0e38f; ki[s] = 0; }
  float wv = 3.0e38f; int wslot = 0;

  const int xlow = clampi(x0 - WR, 0, NW-1);
  // Branch-free norm-staging columns: lane covers m=lane and m=lane+16
  // (m=16..31 written twice with identical values -- harmless).
  const int xn0 = clampi(x0 - WR + lane, 0, NW-1);
  const int xn1 = clampi(x0 - WR + lane + 16, 0, NW-1);

  for (int dyi = 0; dyi < NOFF; ++dyi) {
    int yc = clampi(y + dyi - WR, 0, NH-1);
    const _Float16* brows = Pg + (size_t)((t*NH + yc)*NW) * 32;
    const float*    nrow  = norms + (size_t)((t*NH + yc)*NW);

    // Prefetch next dy's candidate row band (one 128B line per lane ~ 4KB)
    if (dyi + 1 < NOFF) {
      int yn = clampi(y + dyi + 1 - WR, 0, NH-1);
      const _Float16* pf = Pg + ((size_t)((t*NH + yn)*NW) + xlow) * 32;
      __builtin_prefetch(pf + (size_t)lane * 64, 0, 1);
    }

    // Issue candidate-norm loads first, then the six B-fragment loads,
    // so all memory is in flight before the first consumer waits.
    float nc0 = nrow[xn0];
    float nc1 = nrow[xn1];

    v16h bfrag[3];
    #pragma unroll
    for (int tbl = 0; tbl < 3; ++tbl) {
      int m  = tbl*16 + n16;                          // candidate column in Gram
      int xb = clampi(x0 - WR + m, 0, NW-1);          // == reference wn clipping
      const _Float16* brow = brows + (size_t)xb * 32;
      v8h blo = *(const v8h*)(brow + k0);
      v8h bhi = *(const v8h*)(brow + k0 + 16);
      #pragma unroll
      for (int e = 0; e < 8; ++e) { bfrag[tbl][e] = blo[e]; bfrag[tbl][e+8] = bhi[e]; }
    }

    s_nc[lane]      = nc0;
    s_nc[lane + 16] = nc1;

    #pragma unroll
    for (int tbl = 0; tbl < 3; ++tbl) {
      int m = tbl*16 + n16;
      v8f acc = {};
      acc = __builtin_amdgcn_wmma_f32_16x16x32_f16(
          /*neg_a=*/false, afrag, /*neg_b=*/false, bfrag[tbl],
          /*c_mod=*/(short)0, acc, /*reuse_a=*/false, /*reuse_b=*/false);
      // Branch-free D-fragment scatter: lane=N, vgpr=M (hi-lane split)
      #pragma unroll
      for (int v = 0; v < 8; ++v)
        s_gram[(v + hi*8)*GSTRIDE + m] = acc[v];
    }
    __syncthreads();

    if (lane < 16 && (x0 + lane) < NW) {
      float nq = s_nq[lane];
      int oibase = dyi * NOFF;
      const float* grow = &s_gram[lane*GSTRIDE + lane];
      const float* ncrow = &s_nc[lane];
      for (int j = 0; j < NOFF; ++j) {
        float d = nq + ncrow[j] - 2.0f*grow[j];
        if (d < wv) {
          int oi = oibase + j;
          #pragma unroll
          for (int s = 0; s < KNN; ++s) if (s == wslot) { kd[s] = d; ki[s] = oi; }
          wv = kd[0]; wslot = 0;
          #pragma unroll
          for (int s = 1; s < KNN; ++s) if (kd[s] > wv) { wv = kd[s]; wslot = s; }
        }
      }
    }
    __syncthreads();
  }

  if (lane < 16 && (x0 + lane) < NW) {
    size_t q = (size_t)((t*NH + y)*NW) + (x0 + lane);
    #pragma unroll
    for (int s = 0; s < KNN; ++s) { tkD[q*KNN + s] = kd[s]; tkI[q*KNN + s] = ki[s]; }
  }
}

// --------- K4: softmax over 14 neighbors + weighted color gather --------
__global__ __launch_bounds__(96) void k_agg(const float* __restrict__ xp,
                                            const float* __restrict__ tkD,
                                            const int* __restrict__ tkI,
                                            const int* __restrict__ sigma_p,
                                            float* __restrict__ denoS,
                                            float* __restrict__ pwArr) {
  int q = blockIdx.x;
  int t = q / (NH*NW); int r = q % (NH*NW); int y = r / NW; int x = r % NW;
  __shared__ float s_w[KNN];
  __shared__ int   s_hn[KNN], s_wn[KNN];
  __shared__ float s_pw;
  int tid = threadIdx.x;
  if (tid == 0) {
    float d[KNN];
    float dmin = 3.0e38f;
    #pragma unroll
    for (int s = 0; s < KNN; ++s) { d[s] = tkD[(size_t)q*KNN + s]; dmin = fminf(dmin, d[s]); }
    float sig  = ((float)sigma_p[0] / 255.0f) * 2.0f;
    float beta = 1.0f / (2.0f * sig * sig * (float)(PS*PS));
    float e[KNN]; float sum = 0.f;
    #pragma unroll
    for (int s = 0; s < KNN; ++s) { e[s] = __expf(-beta*(d[s]-dmin)); sum += e[s]; }
    #pragma unroll
    for (int s = 0; s < KNN; ++s) s_w[s] = e[s] / sum;
    #pragma unroll
    for (int s = 0; s < KNN; ++s) {
      int oi = tkI[(size_t)q*KNN + s];
      s_hn[s] = clampi(y + oi/NOFF - WR, 0, NH-1);
      s_wn[s] = clampi(x + oi%NOFF - WR, 0, NW-1);
    }
    s_pw = 1.0f / (1.0f + dmin);
  }
  __syncthreads();
  if (tid < CC*PS*PS) {
    int c = tid / 25; int pp = tid % 25; int py = pp / 5; int px = pp % 5;
    float acc = 0.f;
    #pragma unroll
    for (int s = 0; s < KNN; ++s)
      acc += s_w[s] * xp[((size_t)(t*CC + c)*HP + (s_hn[s]+py))*WP + (s_wn[s]+px)];
    denoS[(size_t)q*75 + tid] = acc * s_pw;
  }
  if (tid == 0) pwArr[q] = s_pw;
}

// --------------- K5: 25-tap fold + weighted average + denorm -----------
__global__ void k_fold(const float* __restrict__ denoS, const float* __restrict__ pwArr,
                       const float* __restrict__ means, float* __restrict__ out) {
  int idx = blockIdx.x * blockDim.x + threadIdx.x;
  if (idx >= TT*CC*HH*WW) return;
  int t = idx / (CC*HH*WW); int r = idx % (CC*HH*WW);
  int c = r / (HH*WW); int r2 = r % (HH*WW); int h = r2 / WW; int w = r2 % WW;
  int hp = h + PADW, wp = w + PADW;                   // all 25 taps in-range
  float num = 0.f, den = 0.f;
  #pragma unroll
  for (int dy = 0; dy < PS; ++dy)
    #pragma unroll
    for (int dx = 0; dx < PS; ++dx) {
      int qy = hp - dy, qx = wp - dx;
      size_t q = (size_t)((t*NH + qy)*NW) + qx;
      num += denoS[q*75 + c*25 + dy*5 + dx];
      den += pwArr[q];
    }
  float val = num/den + means[t*CC + c];
  out[idx] = 255.0f * (val*0.5f + 0.5f);
}

extern "C" void kernel_launch(void* const* d_in, const int* in_sizes, int n_in,
                              void* d_out, int out_size, void* d_ws, size_t ws_size,
                              hipStream_t stream) {
  (void)in_sizes; (void)n_in; (void)out_size; (void)ws_size;
  const float* noisy = (const float*)d_in[0];
  const int*   sigma = (const int*)d_in[1];
  float* out = (float*)d_out;

  char* ws = (char*)d_ws;
  size_t off = 0;
  auto alloc = [&](size_t bytes) -> size_t {
    size_t o = off;
    off += (bytes + 255) & ~(size_t)255;
    return o;
  };
  float*    means = (float*)   (ws + alloc(6 * sizeof(float)));
  float*    xp    = (float*)   (ws + alloc((size_t)TT*CC*HP*WP * sizeof(float)));
  float*    gray  = (float*)   (ws + alloc((size_t)TT*HP*WP * sizeof(float)));
  _Float16* Pg    = (_Float16*)(ws + alloc((size_t)NQTOT*32 * sizeof(_Float16)));
  float*    norms = (float*)   (ws + alloc((size_t)NQTOT * sizeof(float)));
  float*    tkD   = (float*)   (ws + alloc((size_t)NQTOT*KNN * sizeof(float)));
  int*      tkI   = (int*)     (ws + alloc((size_t)NQTOT*KNN * sizeof(int)));
  float*    denoS = (float*)   (ws + alloc((size_t)NQTOT*75 * sizeof(float)));
  float*    pwArr = (float*)   (ws + alloc((size_t)NQTOT * sizeof(float)));

  k_means<<<TT*CC, 256, 0, stream>>>(noisy, means);
  int n1 = TT*HP*WP;
  k_pad_gray<<<(n1 + 255)/256, 256, 0, stream>>>(noisy, means, xp, gray);
  k_patches<<<(NQTOT + 255)/256, 256, 0, stream>>>(gray, Pg, norms);
  k_knn<<<TT*NH*TILESX, 32, 0, stream>>>(Pg, norms, tkD, tkI);
  k_agg<<<NQTOT, 96, 0, stream>>>(xp, tkD, tkI, sigma, denoS, pwArr);
  int n5 = TT*CC*HH*WW;
  k_fold<<<(n5 + 255)/256, 256, 0, stream>>>(denoS, pwArr, means, out);
}
